// AUGRUCell_4535485464621
// MI455X (gfx1250) — compile-verified, matrix operand order checked by reading, not run on hardware
//
#include <hip/hip_runtime.h>
#include <hip/hip_bf16.h>

typedef __attribute__((ext_vector_type(16))) __bf16 v16bf;
typedef __attribute__((ext_vector_type(8)))  float  v8f;
typedef int v4i __attribute__((ext_vector_type(4)));
typedef __attribute__((address_space(1))) v4i* gptr4;   // global addrspace
typedef __attribute__((address_space(3))) v4i* lptr4;   // LDS addrspace

#define BDIM         256
#define ROWS_PER_BLK 128
#define ROW_IN       129            // inputs row stride (D_IN + 1 attention col)
#define DH           128
#define WPAD         136            // padded LDS row stride (bf16) -> conflict-free ds_load_b128
#define WELEMS       (128 * WPAD)   // bf16 elems per weight matrix in LDS
#define SMEM_W_BYTES (6 * WELEMS * 2)
#define SMEM_BYTES   (SMEM_W_BYTES + 6 * 128 * 4)
#define WS_BYTES     (6 * 128 * 128 * 2)   // pre-converted bf16 weights in d_ws

#if defined(__has_builtin)
#  if __has_builtin(__builtin_amdgcn_global_load_async_to_lds_b128)
#    define HAVE_ASYNC_LDS 1
#  else
#    define HAVE_ASYNC_LDS 0
#  endif
#else
#  define HAVE_ASYNC_LDS 0
#endif

// ---- B tile (32x16 bf16) from LDS: lane = N, K runs match A layout ----
__device__ __forceinline__ v16bf load_b_tile(const __bf16* __restrict__ w,
                                             int nrow, int kt, int hi) {
  const __bf16* p = w + nrow * WPAD + kt * 32 + hi * 8;
  union { v16bf v; uint4 q[2]; } u;
  u.q[0] = *(const uint4*)(p);        // K = kt*32 + hi*8 .. +8
  u.q[1] = *(const uint4*)(p + 16);   // K = kt*32 + 16 + hi*8 .. +8
  return u.v;
}

// ---- A tile (16x32) from global f32 row, converted to bf16 ----
__device__ __forceinline__ v16bf load_a_f32(const float* __restrict__ rowp,
                                            int kt, int hi) {
  const float* p = rowp + kt * 32 + hi * 8;
  v16bf a;
#pragma unroll
  for (int j = 0; j < 8; ++j) a[j] = (__bf16)p[j];
#pragma unroll
  for (int j = 0; j < 8; ++j) a[8 + j] = (__bf16)p[16 + j];
  return a;
}

// ---- one weight matrix: preload 4 B-tiles, then 4 chained WMMAs per N-tile ----
template<bool INIT>
__device__ __forceinline__ void wmma_mat(v8f acc[4], const v16bf a[4],
                                         const __bf16* __restrict__ wmat,
                                         int nb, int mlane, int hi) {
#pragma unroll
  for (int nt = 0; nt < 4; ++nt) {
    v16bf b[4];
#pragma unroll
    for (int kt = 0; kt < 4; ++kt)
      b[kt] = load_b_tile(wmat, nb + nt * 16 + mlane, kt, hi);
    v8f c;
    if (INIT) { v8f z = {}; c = z; } else { c = acc[nt]; }
#pragma unroll
    for (int kt = 0; kt < 4; ++kt)
      c = __builtin_amdgcn_wmma_f32_16x16x32_bf16(
          false, a[kt], false, b[kt], (short)0, c, false, false);
    acc[nt] = c;
  }
}

__device__ __forceinline__ float sigm(float x) {
  return 1.0f / (1.0f + __expf(-x));
}
__device__ __forceinline__ float tanh_fast(float x) {
  return 1.0f - 2.0f / (__expf(2.0f * x) + 1.0f);
}

// ---- pre-kernel: fp32 weights -> packed bf16 in d_ws (once for whole grid) ----
__global__ __launch_bounds__(BDIM)
void convert_weights_kernel(const float* __restrict__ W0, const float* __restrict__ W1,
                            const float* __restrict__ W2, const float* __restrict__ W3,
                            const float* __restrict__ W4, const float* __restrict__ W5,
                            unsigned short* __restrict__ dst) {
  const int m = blockIdx.y;
  const float* W;
  switch (m) {
    case 0: W = W0; break; case 1: W = W1; break; case 2: W = W2; break;
    case 3: W = W3; break; case 4: W = W4; break; default: W = W5; break;
  }
  const int i = blockIdx.x * BDIM + threadIdx.x;       // float4 index, 4096/matrix
  float4 f = ((const float4*)W)[i];
  union { ushort4 s; uint2 u; } o;
  o.s.x = __builtin_bit_cast(unsigned short, (__bf16)f.x);
  o.s.y = __builtin_bit_cast(unsigned short, (__bf16)f.y);
  o.s.z = __builtin_bit_cast(unsigned short, (__bf16)f.z);
  o.s.w = __builtin_bit_cast(unsigned short, (__bf16)f.w);
  ((uint2*)dst)[m * 4096 + i] = o.u;
}

template<bool USE_WS>
__global__ __launch_bounds__(BDIM)
void augru_wmma_kernel(const unsigned short* __restrict__ wsrc,
                       const float* __restrict__ inputs, const float* __restrict__ h,
                       const float* __restrict__ Wxu, const float* __restrict__ bxu,
                       const float* __restrict__ Whu, const float* __restrict__ bhu,
                       const float* __restrict__ Wxr, const float* __restrict__ bxr,
                       const float* __restrict__ Whr, const float* __restrict__ bhr,
                       const float* __restrict__ Wxg, const float* __restrict__ bxg,
                       const float* __restrict__ Whg, const float* __restrict__ bhg,
                       float* __restrict__ out) {
  extern __shared__ char smem[];
  __bf16* wlds = (__bf16*)smem;
  float*  blds = (float*)(smem + SMEM_W_BYTES);

  const int tid = threadIdx.x;

  // ---- stage weights into padded LDS (bf16) ----
  if (USE_WS) {
    // pre-converted bf16: 16B chunks, async-copy to LDS when available
    const uint4* src = (const uint4*)wsrc;          // 8 bf16 per chunk
    for (int c = tid; c < 6 * 2048; c += BDIM) {
      int m    = c >> 11;
      int rem  = c & 2047;
      int row  = rem >> 4;
      int col8 = rem & 15;
      unsigned ldsoff = (unsigned)(m * WELEMS + row * WPAD + col8 * 8) * 2u;
#if HAVE_ASYNC_LDS
      __builtin_amdgcn_global_load_async_to_lds_b128(
          (gptr4)(src + c), (lptr4)(smem + ldsoff), 0, 0);
#else
      *(uint4*)(smem + ldsoff) = src[c];
#endif
    }
#if HAVE_ASYNC_LDS
#  if __has_builtin(__builtin_amdgcn_s_wait_asynccnt)
    __builtin_amdgcn_s_wait_asynccnt(0);
#  else
    asm volatile("s_wait_asynccnt 0x0" ::: "memory");
#  endif
#endif
  } else {
    // fallback: convert fp32 weights in-kernel
    const float* Ws[6] = {Wxu, Whu, Wxr, Whr, Wxg, Whg};
#pragma unroll
    for (int m = 0; m < 6; ++m) {
      const float4* W4 = (const float4*)Ws[m];
      __bf16* dstm = wlds + m * WELEMS;
      for (int i = tid; i < (128 * 128 / 4); i += BDIM) {
        float4 f = W4[i];
        int row = i >> 5;
        int col = (i & 31) * 4;
        __bf16* d = dstm + row * WPAD + col;
        d[0] = (__bf16)f.x; d[1] = (__bf16)f.y;
        d[2] = (__bf16)f.z; d[3] = (__bf16)f.w;
      }
    }
  }
  {
    const float* Bs[6] = {bxu, bhu, bxr, bhr, bxg, bhg};
#pragma unroll
    for (int m = 0; m < 6; ++m)
      if (tid < 128) blds[m * 128 + tid] = Bs[m][tid];
  }
  __syncthreads();

  const int wave  = tid >> 5;
  const int lane  = tid & 31;
  const int mlane = lane & 15;
  const int hi    = lane >> 4;
  const long r0   = (long)blockIdx.x * ROWS_PER_BLK + wave * 16;

  // ---- A tiles: x (stride 129) and h (stride 128), bf16 16x32 layout ----
  const float* xrow = inputs + (r0 + mlane) * (long)ROW_IN;
  const float* hrow = (const float*)__builtin_assume_aligned(
      h + (r0 + mlane) * (long)DH, 16);
  v16bf ax[4], ah[4];
#pragma unroll
  for (int kt = 0; kt < 4; ++kt) {
    ax[kt] = load_a_f32(xrow, kt, hi);
    ah[kt] = load_a_f32(hrow, kt, hi);
  }

  // attention for this lane's C/D-layout rows: M = e + hi*8
  float att[8];
#pragma unroll
  for (int e = 0; e < 8; ++e)
    att[e] = inputs[(r0 + e + hi * 8) * (long)ROW_IN + 128];

#pragma unroll
  for (int half = 0; half < 2; ++half) {
    const int nb = half * 64;

    // ---- u = sigmoid(x.Wxu^T + h.Whu^T + bxu + bhu) * att ----
    v8f u[4];
    wmma_mat<true >(u, ax, wlds + 0 * WELEMS, nb, mlane, hi);
    wmma_mat<false>(u, ah, wlds + 1 * WELEMS, nb, mlane, hi);
#pragma unroll
    for (int nt = 0; nt < 4; ++nt) {
      float bv = blds[0 * 128 + nb + nt * 16 + mlane] +
                 blds[1 * 128 + nb + nt * 16 + mlane];
#pragma unroll
      for (int e = 0; e < 8; ++e)
        u[nt][e] = sigm(u[nt][e] + bv) * att[e];
    }

    // ---- r = sigmoid(x.Wxr^T + h.Whr^T + bxr + bhr) ----
    v8f r[4];
    wmma_mat<true >(r, ax, wlds + 2 * WELEMS, nb, mlane, hi);
    wmma_mat<false>(r, ah, wlds + 3 * WELEMS, nb, mlane, hi);
#pragma unroll
    for (int nt = 0; nt < 4; ++nt) {
      float bv = blds[2 * 128 + nb + nt * 16 + mlane] +
                 blds[3 * 128 + nb + nt * 16 + mlane];
#pragma unroll
      for (int e = 0; e < 8; ++e)
        r[nt][e] = sigm(r[nt][e] + bv);
    }

    // ---- g = tanh(x.Wxg^T + bxg + r * (h.Whg^T + bhg)) ----
    v8f g[4];
    wmma_mat<true>(g, ah, wlds + 5 * WELEMS, nb, mlane, hi);   // h.Whg^T
#pragma unroll
    for (int nt = 0; nt < 4; ++nt) {
      float bhgv = blds[5 * 128 + nb + nt * 16 + mlane];
      float bxgv = blds[4 * 128 + nb + nt * 16 + mlane];
#pragma unroll
      for (int e = 0; e < 8; ++e)
        g[nt][e] = r[nt][e] * (g[nt][e] + bhgv) + bxgv;
    }
    wmma_mat<false>(g, ax, wlds + 4 * WELEMS, nb, mlane, hi);  // + x.Wxg^T
#pragma unroll
    for (int nt = 0; nt < 4; ++nt)
#pragma unroll
      for (int e = 0; e < 8; ++e)
        g[nt][e] = tanh_fast(g[nt][e]);

    // ---- h_new = h + u*(g - h), C/D layout recombine + store ----
#pragma unroll
    for (int nt = 0; nt < 4; ++nt) {
#pragma unroll
      for (int e = 0; e < 8; ++e) {
        long row = r0 + e + hi * 8;
        long col = nb + nt * 16 + mlane;
        float hv = h[row * (long)DH + col];
        out[row * (long)DH + col] = hv + u[nt][e] * (g[nt][e] - hv);
      }
    }
  }
}

extern "C" void kernel_launch(void* const* d_in, const int* in_sizes, int n_in,
                              void* d_out, int out_size, void* d_ws, size_t ws_size,
                              hipStream_t stream) {
  const float* inputs = (const float*)d_in[0];
  const float* h      = (const float*)d_in[1];
  const float* Wxu = (const float*)d_in[2];
  const float* bxu = (const float*)d_in[3];
  const float* Whu = (const float*)d_in[4];
  const float* bhu = (const float*)d_in[5];
  const float* Wxr = (const float*)d_in[6];
  const float* bxr = (const float*)d_in[7];
  const float* Whr = (const float*)d_in[8];
  const float* bhr = (const float*)d_in[9];
  const float* Wxg = (const float*)d_in[10];
  const float* bxg = (const float*)d_in[11];
  const float* Whg = (const float*)d_in[12];
  const float* bhg = (const float*)d_in[13];
  float* out = (float*)d_out;

  const int nrows = in_sizes[1] / DH;                         // B
  const int nblk  = (nrows + ROWS_PER_BLK - 1) / ROWS_PER_BLK;

  (void)hipFuncSetAttribute((const void*)augru_wmma_kernel<true>,
                            hipFuncAttributeMaxDynamicSharedMemorySize, SMEM_BYTES);
  (void)hipFuncSetAttribute((const void*)augru_wmma_kernel<false>,
                            hipFuncAttributeMaxDynamicSharedMemorySize, SMEM_BYTES);

  if (ws_size >= (size_t)WS_BYTES) {
    unsigned short* wbf = (unsigned short*)d_ws;
    convert_weights_kernel<<<dim3(16, 6), BDIM, 0, stream>>>(
        Wxu, Whu, Wxr, Whr, Wxg, Whg, wbf);
    augru_wmma_kernel<true><<<nblk, BDIM, SMEM_BYTES, stream>>>(
        wbf, inputs, h, Wxu, bxu, Whu, bhu, Wxr, bxr, Whr, bhr,
        Wxg, bxg, Whg, bhg, out);
  } else {
    augru_wmma_kernel<false><<<nblk, BDIM, SMEM_BYTES, stream>>>(
        nullptr, inputs, h, Wxu, bxu, Whu, bhu, Wxr, bxr, Whr, bhr,
        Wxg, bxg, Whg, bhg, out);
  }
}
